// Gate_2903397892758
// MI455X (gfx1250) — compile-verified, matrix operand order checked by reading, not run on hardware
//
#include <hip/hip_runtime.h>
#include <hip/hip_bf16.h>

// ---------------------------------------------------------------------------
// MoE gate (router) for MI455X / gfx1250, wave32 + WMMA bf16.
//   scores = sigmoid(x @ W.T + bias)        [8192 x 256]
//   group top-2 -> top-1 group -> top-8 experts -> renorm * 2.5
// Fused single kernel: GEMM tiles -> scores in LDS -> routing -> (w, idx).
// Software-pipelined: global loads for step i+1 issued during step i's WMMAs.
// d_out layout: w [8192*8] float32, then idx [8192*8] int32 (bit pattern).
// ---------------------------------------------------------------------------

#define M_ROWS   8192
#define KDIM     7168
#define EXPERTS  256
#define N_GROUPS 8
#define EPG      32        // experts per group
#define TOPK     8
#define ROUTE_SCALE 2.5f

#define M_TILE   128       // rows per workgroup
#define KSTEP    32        // bf16 WMMA K depth
#define ASTRIDE  40        // LDS row stride (bf16 elems), padded, 80B = 16B-aligned
#define BSTRIDE  40

typedef __bf16 bf16;
typedef __attribute__((ext_vector_type(16))) __bf16 v16bf;
typedef __attribute__((ext_vector_type(8)))  __bf16 v8bf;
typedef __attribute__((ext_vector_type(4)))  __bf16 v4bf;
typedef __attribute__((ext_vector_type(8)))  float  v8f;
typedef __attribute__((ext_vector_type(4)))  float  v4f;

union V16U { v16bf v; v8bf h[2]; };

__global__ __launch_bounds__(256)
void gate_fused_kernel(const float* __restrict__ x,
                       const float* __restrict__ wgt,
                       const float* __restrict__ bias,
                       float* __restrict__ w_out,
                       int*   __restrict__ idx_out)
{
    __shared__ bf16  lA[M_TILE * ASTRIDE];          // 10240 B
    __shared__ bf16  lB[EXPERTS * BSTRIDE];         // 20480 B
    __shared__ float lS[M_TILE * EXPERTS];          // 131072 B post-sigmoid scores

    const int tid  = threadIdx.x;
    const int lane = tid & 31;
    const int wave = tid >> 5;          // 8 waves
    const int row0 = blockIdx.x * M_TILE;

    const int hsel = (lane >> 4) & 1;   // hi-half lane select (ISA 16-bit A/B layout)
    const int cl   = lane & 15;
    const int n0   = wave * 32;         // this wave's 32 expert columns

    // Per-thread staging slots (constant across K): 4 float4 of A, 8 float4 of B.
    const int ar  = tid >> 3;                    // A rows: tid/8 + {0,32,64,96}
    const int ac4 = (tid & 7) << 2;              // A k offset 0..28
    const int be  = tid >> 3;                    // B experts: tid/8 + {0,32,..,224}
    const int bc4 = (tid & 7) << 2;

    // 8 M-subtiles x 2 N-subtiles of 16x16 f32 accumulators = 128 VGPRs
    v8f acc[8][2];
#pragma unroll
    for (int m = 0; m < 8; ++m)
#pragma unroll
        for (int t = 0; t < 2; ++t)
            acc[m][t] = (v8f)0.0f;

    // ---- prologue: prefetch k-step 0 into registers
    v4f pa[4], pb[8];
#pragma unroll
    for (int i = 0; i < 4; ++i)
        pa[i] = __builtin_nontemporal_load(
            (const v4f*)(x + (size_t)(row0 + ar + 32 * i) * KDIM + ac4));
#pragma unroll
    for (int i = 0; i < 8; ++i)
        pb[i] = *(const v4f*)(wgt + (size_t)(be + 32 * i) * KDIM + bc4);

    for (int k0 = 0; k0 < KDIM; k0 += KSTEP) {
        // ---- commit prefetched step to LDS (fp32 -> bf16)
#pragma unroll
        for (int i = 0; i < 4; ++i) {
            v4bf b;
            b.x = (bf16)pa[i].x; b.y = (bf16)pa[i].y;
            b.z = (bf16)pa[i].z; b.w = (bf16)pa[i].w;
            *(v4bf*)(&lA[(ar + 32 * i) * ASTRIDE + ac4]) = b;
        }
#pragma unroll
        for (int i = 0; i < 8; ++i) {
            v4bf b;
            b.x = (bf16)pb[i].x; b.y = (bf16)pb[i].y;
            b.z = (bf16)pb[i].z; b.w = (bf16)pb[i].w;
            *(v4bf*)(&lB[(be + 32 * i) * BSTRIDE + bc4]) = b;
        }
        __syncthreads();

        // ---- issue global prefetch for step k0+KSTEP (overlaps WMMAs below).
        //      Last iteration re-reads step 0 (harmless, keeps code uniform).
        {
            int kn = (k0 + KSTEP < KDIM) ? (k0 + KSTEP) : 0;
#pragma unroll
            for (int i = 0; i < 4; ++i)
                pa[i] = __builtin_nontemporal_load(
                    (const v4f*)(x + (size_t)(row0 + ar + 32 * i) * KDIM + kn + ac4));
#pragma unroll
            for (int i = 0; i < 8; ++i)
                pb[i] = *(const v4f*)(wgt + (size_t)(be + 32 * i) * KDIM + kn + bc4);
        }

        // ---- B fragments for this wave's two N tiles (ISA 16-bit B layout:
        //      lane<16: K {0..7,16..23}; lane>=16: K {8..15,24..31})
        V16U b0, b1;
        {
            const bf16* p0 = &lB[(n0 + cl) * BSTRIDE + hsel * 8];
            b0.h[0] = *(const v8bf*)(p0);
            b0.h[1] = *(const v8bf*)(p0 + 16);
            const bf16* p1 = &lB[(n0 + 16 + cl) * BSTRIDE + hsel * 8];
            b1.h[0] = *(const v8bf*)(p1);
            b1.h[1] = *(const v8bf*)(p1 + 16);
        }

        // ---- 8 M-subtiles: load A fragment once, feed both N tiles
#pragma unroll
        for (int m = 0; m < 8; ++m) {
            V16U a;
            const bf16* p = &lA[(m * 16 + cl) * ASTRIDE + hsel * 8];
            a.h[0] = *(const v8bf*)(p);
            a.h[1] = *(const v8bf*)(p + 16);
            acc[m][0] = __builtin_amdgcn_wmma_f32_16x16x32_bf16(
                false, a.v, false, b0.v, (short)0, acc[m][0], false, false);
            acc[m][1] = __builtin_amdgcn_wmma_f32_16x16x32_bf16(
                false, a.v, false, b1.v, (short)0, acc[m][1], false, false);
        }
        __syncthreads();
    }

    // ---- epilogue: bias + sigmoid, scatter to score LDS per C-layout
    //      (VGPR r: lanes 0-15 -> M=r, lanes 16-31 -> M=8+r; N = lane&15)
#pragma unroll
    for (int m = 0; m < 8; ++m)
#pragma unroll
        for (int t = 0; t < 2; ++t) {
            int col = n0 + t * 16 + cl;
            float bv = bias[col];
#pragma unroll
            for (int r = 0; r < 8; ++r) {
                int rl = m * 16 + hsel * 8 + r;
                float z = acc[m][t][r] + bv;
                lS[rl * EXPERTS + col] = 1.0f / (1.0f + __expf(-z));
            }
        }
    __syncthreads();

    // ---- routing: one thread per row
    if (tid < M_TILE) {
        const float* s = &lS[tid * EXPERTS];

        // group score = top1 + top2 within group; pick best group (tie -> lowest idx)
        int bestg = 0; float bestgs = -1e30f;
#pragma unroll
        for (int g = 0; g < N_GROUPS; ++g) {
            float m1 = -1e30f, m2 = -1e30f;
            for (int e = 0; e < EPG; ++e) {
                float v = s[g * EPG + e];
                if (v > m1) { m2 = m1; m1 = v; }
                else if (v > m2) { m2 = v; }
            }
            float gs = m1 + m2;
            if (gs > bestgs) { bestgs = gs; bestg = g; }
        }

        // top-8 within the winning group's 32 experts (desc, tie -> lowest idx)
        float loc[EPG];
#pragma unroll
        for (int e = 0; e < EPG; ++e) loc[e] = s[bestg * EPG + e];

        float wsel[TOPK]; int isel[TOPK]; float sum = 0.0f;
#pragma unroll
        for (int k = 0; k < TOPK; ++k) {
            int bi = 0; float bv = -1e30f;
            for (int e = 0; e < EPG; ++e)
                if (loc[e] > bv) { bv = loc[e]; bi = e; }
            wsel[k] = bv;
            isel[k] = bestg * EPG + bi;
            loc[bi] = -1e30f;
            sum += bv;
        }

        float scale = ROUTE_SCALE / sum;
        long long grow = (long long)row0 + tid;
#pragma unroll
        for (int k = 0; k < TOPK; ++k) {
            w_out[grow * TOPK + k]  = wsel[k] * scale;
            idx_out[grow * TOPK + k] = isel[k];
        }
    }
}

extern "C" void kernel_launch(void* const* d_in, const int* in_sizes, int n_in,
                              void* d_out, int out_size, void* d_ws, size_t ws_size,
                              hipStream_t stream) {
    (void)in_sizes; (void)n_in; (void)d_ws; (void)ws_size; (void)out_size;
    const float* x    = (const float*)d_in[0];
    const float* wgt  = (const float*)d_in[1];
    const float* bias = (const float*)d_in[2];
    float* w_out  = (float*)d_out;
    int*   idx_out = (int*)((float*)d_out + (size_t)M_ROWS * TOPK);

    dim3 grid(M_ROWS / M_TILE);   // 64 workgroups
    dim3 block(256);              // 8 wave32 waves
    gate_fused_kernel<<<grid, block, 0, stream>>>(x, wgt, bias, w_out, idx_out);
}
